// StandardAttention_72559177498893
// MI455X (gfx1250) — matrix operand order repaired and checked by reading, not compiled, not run on hardware
//
#include <hip/hip_runtime.h>
#include <hip/hip_bf16.h>

typedef __attribute__((ext_vector_type(8)))  __bf16 v8bf;
typedef __attribute__((ext_vector_type(16))) __bf16 v16bf;
typedef __attribute__((ext_vector_type(8)))  float  v8f;

// ---------------------------------------------------------------------------
// WMMA helpers (V_WMMA_F32_16X16X32_BF16, wave32)
//
// A fragment 16x32 bf16 (row-major source, leading dim lda):
//   lane<16 : row = lane,     v[0..7]=K[0..7],   v[8..15]=K[16..23]
//   lane>=16: row = lane-16,  v[0..7]=K[8..15],  v[8..15]=K[24..31]
// B fragment 32x16 bf16, loaded from BT[N][K] (= B transposed, row-major):
//   lane<16 : col = lane,     v[0..7]=K[0..7],   v[8..15]=K[16..23]
//   lane>=16: col = lane-16,  v[0..7]=K[8..15],  v[8..15]=K[24..31]
// C/D 16x16 f32: lane<16: col=lane, v[r]=row r (0..7); lane>=16: rows 8..15.
// ---------------------------------------------------------------------------

__device__ __forceinline__ v8f zero8() {
  v8f z;
#pragma unroll
  for (int i = 0; i < 8; ++i) z[i] = 0.0f;
  return z;
}

__device__ __forceinline__ v16bf load_a32(const __bf16* A, int lda, int lane) {
  int row  = lane & 15;
  int koff = (lane & 16) ? 8 : 0;
  const __bf16* p = A + (size_t)row * lda + koff;
  v8bf lo = *(const v8bf*)(p);       // K = koff .. koff+7
  v8bf hi = *(const v8bf*)(p + 16);  // K = koff+16 .. koff+23
  v16bf r;
#pragma unroll
  for (int i = 0; i < 8; ++i) { r[i] = lo[i]; r[i + 8] = hi[i]; }
  return r;
}

__device__ __forceinline__ v16bf load_bt32(const __bf16* BT, int ldbt, int lane) {
  int col = lane & 15;
  int kb  = (lane & 16) ? 8 : 0;
  const __bf16* p = BT + (size_t)col * ldbt + kb;
  v8bf lo = *(const v8bf*)(p);
  v8bf hi = *(const v8bf*)(p + 16);
  v16bf r;
#pragma unroll
  for (int i = 0; i < 8; ++i) { r[i] = lo[i]; r[i + 8] = hi[i]; }
  return r;
}

__device__ __forceinline__ v8f wmma_bf16(v16bf a, v16bf b, v8f c) {
  return __builtin_amdgcn_wmma_f32_16x16x32_bf16(false, a, false, b, (short)0, c,
                                                 false, false);
}

// ---------------------------------------------------------------------------
// Elementwise conversion kernels
// ---------------------------------------------------------------------------

__global__ void cvt_bf16_kernel(const float* __restrict__ in,
                                __bf16* __restrict__ out, int n) {
  int i = blockIdx.x * blockDim.x + threadIdx.x;
  if (i < n) out[i] = (__bf16)in[i];
}

// W[K][N] f32 -> WT[N][K] bf16
__global__ void transpose_cvt_kernel(const float* __restrict__ in,
                                     __bf16* __restrict__ out, int K, int N) {
  int i = blockIdx.x * blockDim.x + threadIdx.x;
  if (i >= K * N) return;
  int n = i % N, k = i / N;
  out[(size_t)n * K + k] = (__bf16)in[(size_t)k * N + n];
}

// in: f32 [B,S,NHD*128] (GEMM layout), out: bf16 [B,NHD,S,128] with RoPE
__global__ void rope_cvt_kernel(const float* __restrict__ in,
                                const float* __restrict__ cosT,
                                const float* __restrict__ sinT,
                                __bf16* __restrict__ out, int B, int S, int NHD) {
  const int HD = 128;
  int i = blockIdx.x * blockDim.x + threadIdx.x;
  int total = B * NHD * S * 64;
  if (i >= total) return;
  int d2 = i & 63;
  int s  = (i >> 6) % S;
  int h  = ((i >> 6) / S) % NHD;
  int b  = i / (64 * S * NHD);
  size_t ibase = ((size_t)(b * S + s)) * (NHD * HD) + (size_t)h * HD;
  float x1 = in[ibase + d2];
  float x2 = in[ibase + d2 + 64];
  float c  = cosT[s * 64 + d2];
  float sn = sinT[s * 64 + d2];
  size_t obase = (((size_t)b * NHD + h) * S + s) * HD;
  out[obase + d2]      = (__bf16)(x1 * c - x2 * sn);
  out[obase + d2 + 64] = (__bf16)(x2 * c + x1 * sn);
}

// Vf: f32 [B,S,4*128] -> Vt: bf16 [B,4,128,S]
__global__ void vtrans_kernel(const float* __restrict__ Vf,
                              __bf16* __restrict__ Vt, int B, int S) {
  const int NKV = 4, HD = 128;
  int i = blockIdx.x * blockDim.x + threadIdx.x;
  int total = B * NKV * HD * S;
  if (i >= total) return;
  int s   = i % S;
  int d   = (i / S) % HD;
  int kvh = (i / (S * HD)) % NKV;
  int b   = i / (S * HD * NKV);
  Vt[i] = (__bf16)Vf[((size_t)(b * S + s)) * (NKV * HD) + kvh * HD + d];
}

// ---------------------------------------------------------------------------
// GEMM: C[M][N] f32 = A[M][K] bf16 * BT[N][K] bf16.
// 8 waves/block arranged 2(M) x 4(N); wave tile 64x64 = 4x4 WMMA frags
// (128 VGPRs of accumulators; legal under wave32's 1024-VGPR budget).
// Per K-step: 16 b128 loads : 16 v_wmma (1:1), 4x operand reuse per fragment.
// Block tile 128(M) x 256(N). M%128==0, N%256==0, K%32==0 assumed.
// ---------------------------------------------------------------------------
__global__ void __launch_bounds__(256) gemm_bf16_kernel(
    const __bf16* __restrict__ A, const __bf16* __restrict__ BT,
    float* __restrict__ C, int M, int N, int K) {
  int w = threadIdx.x >> 5, lane = threadIdx.x & 31;
  int wm = w & 1, wn = w >> 1;
  int m0 = blockIdx.x * 128 + wm * 64;
  int n0 = blockIdx.y * 256 + wn * 64;

  v8f acc[4][4];
#pragma unroll
  for (int i = 0; i < 4; ++i)
#pragma unroll
    for (int j = 0; j < 4; ++j) acc[i][j] = zero8();

  for (int k0 = 0; k0 < K; k0 += 32) {
    v16bf bf[4];
#pragma unroll
    for (int j = 0; j < 4; ++j)
      bf[j] = load_bt32(BT + (size_t)(n0 + 16 * j) * K + k0, K, lane);
#pragma unroll
    for (int i = 0; i < 4; ++i) {
      v16bf af = load_a32(A + (size_t)(m0 + 16 * i) * K + k0, K, lane);
#pragma unroll
      for (int j = 0; j < 4; ++j) acc[i][j] = wmma_bf16(af, bf[j], acc[i][j]);
    }
  }

  int rbase = (lane & 16) ? 8 : 0;
  int col   = lane & 15;
#pragma unroll
  for (int i = 0; i < 4; ++i)
#pragma unroll
    for (int j = 0; j < 4; ++j)
#pragma unroll
      for (int r = 0; r < 8; ++r)
        C[(size_t)(m0 + i * 16 + rbase + r) * N + (n0 + j * 16 + col)] =
            acc[i][j][r];
}

// ---------------------------------------------------------------------------
// Flash attention (causal, GQA 16Q/4KV, hd=128), one wave per 16-row Q tile.
// Q: bf16 [B,16,S,128]  K: bf16 [B,4,S,128]  Vt: bf16 [B,4,128,S]
// O: bf16 [B,S,16*128]
// ---------------------------------------------------------------------------
__global__ void __launch_bounds__(128) flash_attn_kernel(
    const __bf16* __restrict__ Q, const __bf16* __restrict__ Kb,
    const __bf16* __restrict__ Vt, __bf16* __restrict__ O, int S) {
  const int NH = 16, NKV = 4, HD = 128;
  __shared__ __bf16 pbuf[4][16 * 32];  // per-wave P tile staging (C->A relayout)

  int w = threadIdx.x >> 5, lane = threadIdx.x & 31;
  int tiles = S / 64;
  int bh = blockIdx.x / tiles;
  int t  = blockIdx.x % tiles;
  int b = bh / NH, h = bh % NH, kvh = h / (NH / NKV);
  int q0 = t * 64 + w * 16;

  const __bf16* Qh = Q + (((size_t)b * NH + h) * S + q0) * HD;
  const __bf16* Kh = Kb + ((size_t)b * NKV + kvh) * (size_t)S * HD;
  const __bf16* Vh = Vt + ((size_t)b * NKV + kvh) * (size_t)HD * S;

  v16bf qa[4];
#pragma unroll
  for (int kd = 0; kd < 4; ++kd) qa[kd] = load_a32(Qh + kd * 32, HD, lane);

  v8f o[8];
  float m[8], l[8];
#pragma unroll
  for (int j = 0; j < 8; ++j) o[j] = zero8();
#pragma unroll
  for (int r = 0; r < 8; ++r) { m[r] = -3.0e38f; l[r] = 0.0f; }

  const float scale = 0.08838834764831845f;  // 1/sqrt(128)
  int rbase = (lane & 16) ? 8 : 0;
  int colL  = lane & 15;
  int ktmax = (q0 + 15) >> 5;  // causal upper bound (wave-uniform)

  for (int kt = 0; kt <= ktmax; ++kt) {
    int kv0 = kt * 32;
    // --- scores: S = Q * K^T (16 x 32) ---
    v8f s0 = zero8(), s1 = zero8();
#pragma unroll
    for (int kd = 0; kd < 4; ++kd) {
      v16bf b0 = load_bt32(Kh + (size_t)kv0 * HD + kd * 32, HD, lane);
      v16bf b1 = load_bt32(Kh + (size_t)(kv0 + 16) * HD + kd * 32, HD, lane);
      s0 = wmma_bf16(qa[kd], b0, s0);
      s1 = wmma_bf16(qa[kd], b1, s1);
    }
    // --- causal mask + online softmax (row stats live in a 16-lane half) ---
#pragma unroll
    for (int r = 0; r < 8; ++r) {
      int rowg = q0 + rbase + r;
      float e0 = s0[r] * scale; if (kv0 + colL      > rowg) e0 = -1.0e30f;
      float e1 = s1[r] * scale; if (kv0 + 16 + colL > rowg) e1 = -1.0e30f;
      float vmax = fmaxf(e0, e1);
#pragma unroll
      for (int off = 1; off < 16; off <<= 1)
        vmax = fmaxf(vmax, __shfl_xor(vmax, off, 16));
      float mn    = fmaxf(m[r], vmax);
      float alpha = __expf(m[r] - mn);
      float p0 = __expf(e0 - mn), p1 = __expf(e1 - mn);
      float rs = p0 + p1;
#pragma unroll
      for (int off = 1; off < 16; off <<= 1) rs += __shfl_xor(rs, off, 16);
      m[r] = mn;
      l[r] = l[r] * alpha + rs;
#pragma unroll
      for (int j = 0; j < 8; ++j) o[j][r] *= alpha;
      pbuf[w][(rbase + r) * 32 + colL]      = (__bf16)p0;
      pbuf[w][(rbase + r) * 32 + colL + 16] = (__bf16)p1;
    }
    // same-wave LDS RAW: DS ops are in-order per wave; fence the compiler + HW
    asm volatile("s_wait_dscnt 0" ::: "memory");
    v16bf pa = load_a32(&pbuf[w][0], 32, lane);
    // --- O += P * V ---
#pragma unroll
    for (int jn = 0; jn < 8; ++jn) {
      v16bf bv = load_bt32(Vh + (size_t)(jn * 16) * S + kv0, S, lane);
      o[jn] = wmma_bf16(pa, bv, o[jn]);
    }
  }

  // epilogue: normalize, store bf16 into [B,S,NH*HD]
#pragma unroll
  for (int r = 0; r < 8; ++r) {
    float inv = 1.0f / l[r];
    int rowg = q0 + rbase + r;
    size_t obase = ((size_t)b * S + rowg) * (NH * HD) + (size_t)h * HD;
#pragma unroll
    for (int jn = 0; jn < 8; ++jn)
      O[obase + jn * 16 + colL] = (__bf16)(o[jn][r] * inv);
  }
}

// ---------------------------------------------------------------------------
// Orchestration
// ---------------------------------------------------------------------------
extern "C" void kernel_launch(void* const* d_in, const int* in_sizes, int n_in,
                              void* d_out, int out_size, void* d_ws, size_t ws_size,
                              hipStream_t stream) {
  const int B = 2, S = 2048, H = 2048, NH = 16, NKV = 4, HD = 128;
  const int KV = NKV * HD;  // 512
  const int MS = B * S;     // 4096 rows

  const float* x  = (const float*)d_in[0];
  const float* rc = (const float*)d_in[1];
  const float* rs = (const float*)d_in[2];
  // d_in[3] = attention_mask (causal; applied analytically)
  const float* wq = (const float*)d_in[4];
  const float* wk = (const float*)d_in[5];
  const float* wv = (const float*)d_in[6];
  const float* wo = (const float*)d_in[7];
  float* out = (float*)d_out;

  char* ws = (char*)d_ws;
  size_t off = 0;
  auto alloc = [&](size_t bytes) -> void* {
    void* p = ws + off;
    off += (bytes + 255) & ~(size_t)255;
    return p;
  };
  __bf16* xb  = (__bf16*)alloc((size_t)MS * H * 2);   // 16 MB
  __bf16* wqT = (__bf16*)alloc((size_t)H * H * 2);    //  8 MB
  __bf16* wkT = (__bf16*)alloc((size_t)H * KV * 2);   //  2 MB
  __bf16* wvT = (__bf16*)alloc((size_t)H * KV * 2);   //  2 MB
  __bf16* woT = (__bf16*)alloc((size_t)H * H * 2);    //  8 MB
  float*  Qf  = (float*)alloc((size_t)MS * H * 4);    // 32 MB (reused as Ob)
  float*  Kf  = (float*)alloc((size_t)MS * KV * 4);   //  8 MB
  float*  Vf  = (float*)alloc((size_t)MS * KV * 4);   //  8 MB
  __bf16* Qb  = (__bf16*)alloc((size_t)MS * H * 2);   // 16 MB
  __bf16* Kb  = (__bf16*)alloc((size_t)MS * KV * 2);  //  4 MB
  __bf16* Vt  = (__bf16*)alloc((size_t)MS * KV * 2);  //  4 MB
  __bf16* Ob  = (__bf16*)Qf;  // Qf is dead after RoPE; alias for attn output

  // 1) bf16 conversions / transposes
  cvt_bf16_kernel<<<(MS * H + 255) / 256, 256, 0, stream>>>(x, xb, MS * H);
  transpose_cvt_kernel<<<(H * H + 255) / 256, 256, 0, stream>>>(wq, wqT, H, H);
  transpose_cvt_kernel<<<(H * KV + 255) / 256, 256, 0, stream>>>(wk, wkT, H, KV);
  transpose_cvt_kernel<<<(H * KV + 255) / 256, 256, 0, stream>>>(wv, wvT, H, KV);
  transpose_cvt_kernel<<<(H * H + 255) / 256, 256, 0, stream>>>(wo, woT, H, H);

  // 2) projections (WMMA GEMMs); block tile 128x256
  dim3 gq(MS / 128, H / 256);
  gemm_bf16_kernel<<<gq, 256, 0, stream>>>(xb, wqT, Qf, MS, H, H);
  dim3 gk(MS / 128, KV / 256);
  gemm_bf16_kernel<<<gk, 256, 0, stream>>>(xb, wkT, Kf, MS, KV, H);
  gemm_bf16_kernel<<<gk, 256, 0, stream>>>(xb, wvT, Vf, MS, KV, H);

  // 3) RoPE + layout to head-major bf16; V transpose
  rope_cvt_kernel<<<(B * NH * S * 64 + 255) / 256, 256, 0, stream>>>(
      Qf, rc, rs, Qb, B, S, NH);
  rope_cvt_kernel<<<(B * NKV * S * 64 + 255) / 256, 256, 0, stream>>>(
      Kf, rc, rs, Kb, B, S, NKV);
  vtrans_kernel<<<(B * NKV * HD * S + 255) / 256, 256, 0, stream>>>(Vf, Vt, B, S);

  // 4) causal GQA flash attention (WMMA)
  flash_attn_kernel<<<B * NH * (S / 64), 128, 0, stream>>>(Qb, Kb, Vt, Ob, S);

  // 5) output projection -> f32 out
  dim3 go(MS / 128, H / 256);
  gemm_bf16_kernel<<<go, 256, 0, stream>>>(Ob, woT, out, MS, H, H);

  (void)in_sizes; (void)n_in; (void)out_size; (void)ws_size;
}
